// RNNModel_55155970015295
// MI455X (gfx1250) — matrix operand order repaired and checked by reading, not compiled, not run on hardware
//
#include <hip/hip_runtime.h>

typedef __attribute__((ext_vector_type(16))) _Float16 v16h;
typedef __attribute__((ext_vector_type(8)))  float    v8f;

#define B_  1024
#define T1  512
#define T2  64
#define TT  576
#define DF  9
#define H_  100
#define G4  400      // 4*H
#define NT  25       // N tiles of 16 covering 400
#define KTL 4        // K tiles of 32 covering 128 (H=100 zero-padded)
#define ZS  408      // zbuf row stride (floats)
#define HP  104      // hplain row stride (floats)
#define NTHREADS 512 // 16 wave32

__device__ __forceinline__ float fast_sigmoid(float x) {
    return __builtin_amdgcn_rcpf(1.0f + __expf(-x));
}
__device__ __forceinline__ float fast_tanh(float x) {
    return 2.0f * __builtin_amdgcn_rcpf(1.0f + __expf(-2.0f * x)) - 1.0f;
}

// One workgroup (512 threads = 16 wave32) owns 16 batch rows and runs the whole
// 576-step recurrence locally. Each wave owns <=2 N-tiles of W_h, held as WMMA
// B fragments in 64 register-resident VGPRs (constant indices only -> no spill).
// h lives in LDS pre-swizzled into the 16-bit WMMA A layout; c lives in VGPRs.
__global__ __launch_bounds__(NTHREADS)
void lstm_fused_kernel(const float* __restrict__ input,
                       const float* __restrict__ task,
                       const float* __restrict__ W_in,  const float* __restrict__ b_in,
                       const float* __restrict__ W_task,const float* __restrict__ b_task,
                       const float* __restrict__ W_x,   const float* __restrict__ W_h,
                       const float* __restrict__ b_lstm,
                       const float* __restrict__ W_out, const float* __restrict__ b_out,
                       float* __restrict__ out)
{
    __shared__ float sWx[DF * G4];                 // 14.4 KB
    __shared__ float sWin[DF * DF], sWtk[DF * DF];
    __shared__ float sbin[DF], sbtk[DF], sbout[DF];
    __shared__ float sbl[G4];
    __shared__ float sWout[H_ * DF];               // 3.6 KB
    __shared__ float zbuf[16 * ZS];                // 26.1 KB  (z = zx + h@W_h)
    __shared__ __attribute__((aligned(32))) _Float16 hswz[KTL * 32 * 16]; // 4 KB, A-frag layout
    __shared__ float enc[16 * 12];
    __shared__ float hplain[16 * HP];              // 6.6 KB
    __shared__ float slog[16 * DF];

    const int tid  = threadIdx.x;
    const int wave = tid >> 5;
    const int lane = tid & 31;
    const int hi   = lane >> 4;
    const int ln   = lane & 15;
    const int row0 = blockIdx.x * 16;
    // scalar (SGPR) copy of the wave id -> uniform branches, EXEC untouched
    const int wave_u = __builtin_amdgcn_readfirstlane(wave);

    // ---- one-time: weights to LDS ----
    for (int i = tid; i < DF * G4; i += NTHREADS) sWx[i] = W_x[i];
    for (int i = tid; i < DF * DF; i += NTHREADS) { sWin[i] = W_in[i]; sWtk[i] = W_task[i]; }
    if (tid < DF) { sbin[tid] = b_in[tid]; sbtk[tid] = b_task[tid]; sbout[tid] = b_out[tid]; }
    for (int i = tid; i < G4; i += NTHREADS) sbl[i] = b_lstm[i];
    for (int i = tid; i < H_ * DF; i += NTHREADS) sWout[i] = W_out[i];
    for (int i = tid; i < KTL * 32 * 16; i += NTHREADS) hswz[i] = (_Float16)0.0f; // h0=0 (+K pad)

    // ---- one-time: W_h -> register-resident B fragments (f16) ----
    // B frag (KxN), this code's k order, same as A: lane L holds column N=L&15; K = (L>>4)*16 + j for half j.
    // Slot 0: nt = wave (always < 25). Slot 1: nt = wave+16 (valid iff wave < 9).
    v16h bfrag[KTL][2];
#pragma unroll
    for (int s = 0; s < 2; ++s) {
        const int nt = wave + 16 * s;
#pragma unroll
        for (int kt = 0; kt < KTL; ++kt) {
            v16h f;
#pragma unroll
            for (int j = 0; j < 16; ++j) {
                const int K = kt * 32 + hi * 16 + j;
                const int N = nt * 16 + ln;
                const float v = (nt < NT && K < H_) ? W_h[K * G4 + N] : 0.0f;
                f[j] = (_Float16)v;
            }
            bfrag[kt][s] = f;
        }
    }

    float creg[4];                                  // c state: entry e = tid + 512k -> (row=e&15, u=e>>4)
#pragma unroll
    for (int k = 0; k < 4; ++k) creg[k] = 0.0f;
    __syncthreads();

    for (int t = 0; t < TT; ++t) {
        // ---- A: 9x9 encoders (relu), 144 threads ----
        if (tid < 16 * DF) {
            const int r = tid / DF, d = tid % DF;
            const float* raw; const float* W; const float* bb;
            if (t < T1) { raw = input + ((size_t)(row0 + r) * T1 + t) * DF;        W = sWin; bb = sbin; }
            else        { raw = task  + ((size_t)(row0 + r) * T2 + (t - T1)) * DF; W = sWtk; bb = sbtk; }
            float a = bb[d];
#pragma unroll
            for (int f = 0; f < DF; ++f) a += raw[f] * W[f * DF + d];
            enc[r * 12 + d] = fmaxf(a, 0.0f);
            if (t + 1 < TT) {
                const float* nraw = (t + 1 < T1)
                    ? input + ((size_t)(row0 + r) * T1 + (t + 1)) * DF
                    : task  + ((size_t)(row0 + r) * T2 + (t + 1 - T1)) * DF;
                __builtin_prefetch(nraw, 0, 0);     // global_prefetch_b8 for next step
            }
        }
        __syncthreads();

        // ---- B: z_pre = enc @ W_x + b_lstm  (recomputed; avoids 1.9 GB of HBM zx traffic) ----
        for (int i = tid; i < 16 * G4; i += NTHREADS) {
            const int col = i % G4, r = i / G4;
            float a = sbl[col];
#pragma unroll
            for (int f = 0; f < DF; ++f) a += enc[r * 12 + f] * sWx[f * G4 + col];
            zbuf[r * ZS + col] = a;
        }
        __syncthreads();

        // ---- C: z += h @ W_h via v_wmma_f32_16x16x32_f16 (straight-line, constant bfrag idx) ----
        {
            const v16h* hv = (const v16h*)hswz;     // 32B contiguous per lane per K-tile
            const v16h a0 = hv[0 * 32 + lane];
            const v16h a1 = hv[1 * 32 + lane];
            const v16h a2 = hv[2 * 32 + lane];
            const v16h a3 = hv[3 * 32 + lane];

            // slot 0: nt = wave (always valid)
            {
                const int N = wave * 16 + ln;
                v8f acc;
#pragma unroll
                for (int r = 0; r < 8; ++r) acc[r] = zbuf[(r + 8 * hi) * ZS + N];
                acc = __builtin_amdgcn_wmma_f32_16x16x32_f16(false, a0, false, bfrag[0][0], (short)0, acc, false, false);
                acc = __builtin_amdgcn_wmma_f32_16x16x32_f16(false, a1, false, bfrag[1][0], (short)0, acc, false, false);
                acc = __builtin_amdgcn_wmma_f32_16x16x32_f16(false, a2, false, bfrag[2][0], (short)0, acc, false, false);
                acc = __builtin_amdgcn_wmma_f32_16x16x32_f16(false, a3, false, bfrag[3][0], (short)0, acc, false, false);
#pragma unroll
                for (int r = 0; r < 8; ++r) zbuf[(r + 8 * hi) * ZS + N] = acc[r];
            }
            // slot 1: nt = wave+16 (valid for waves 0..8) -- scalar branch, EXEC stays full
            if (wave_u < 9) {
                const int N = (wave + 16) * 16 + ln;
                v8f acc;
#pragma unroll
                for (int r = 0; r < 8; ++r) acc[r] = zbuf[(r + 8 * hi) * ZS + N];
                acc = __builtin_amdgcn_wmma_f32_16x16x32_f16(false, a0, false, bfrag[0][1], (short)0, acc, false, false);
                acc = __builtin_amdgcn_wmma_f32_16x16x32_f16(false, a1, false, bfrag[1][1], (short)0, acc, false, false);
                acc = __builtin_amdgcn_wmma_f32_16x16x32_f16(false, a2, false, bfrag[2][1], (short)0, acc, false, false);
                acc = __builtin_amdgcn_wmma_f32_16x16x32_f16(false, a3, false, bfrag[3][1], (short)0, acc, false, false);
#pragma unroll
                for (int r = 0; r < 8; ++r) zbuf[(r + 8 * hi) * ZS + N] = acc[r];
            }
        }
        __syncthreads();

        // ---- D: gates (Keras order i,f,g,o), update c (regs), emit h (swizzled f16 + plain f32) ----
#pragma unroll
        for (int k = 0; k < 4; ++k) {
            const int e = tid + NTHREADS * k;
            if (e < 16 * H_) {
                const int r = e & 15, u = e >> 4;
                const float zi = zbuf[r * ZS + u];
                const float zf = zbuf[r * ZS + 100 + u];
                const float zg = zbuf[r * ZS + 200 + u];
                const float zo = zbuf[r * ZS + 300 + u];
                const float ig = fast_sigmoid(zi);
                const float fg = fast_sigmoid(zf);
                const float gg = fast_tanh(zg);
                const float og = fast_sigmoid(zo);
                const float c  = fg * creg[k] + ig * gg;
                creg[k] = c;
                const float h = og * fast_tanh(c);
                hplain[r * HP + u] = h;
                // invert 16-bit A-fragment layout: K-tile kt, in-tile ko
                const int kt = u >> 5, ko = u & 31;
                const int h2 = ko >> 4;
                const int j  = ko & 15;
                hswz[(kt * 32 + r + 16 * h2) * 16 + j] = (_Float16)h;
            }
        }
        __syncthreads();

        // ---- E: logits = h @ W_out + b_out, 144 threads ----
        if (tid < 16 * DF) {
            const int r = tid / DF, d = tid % DF;
            float a = sbout[d];
            for (int u = 0; u < H_; ++u) a += hplain[r * HP + u] * sWout[u * DF + d];
            slog[r * DF + d] = a;
        }
        __syncthreads();

        // ---- F: softmax + store ----
        if (tid < 16) {
            const int r = tid;
            float m = slog[r * DF];
#pragma unroll
            for (int d = 1; d < DF; ++d) m = fmaxf(m, slog[r * DF + d]);
            float ex[DF]; float sum = 0.0f;
#pragma unroll
            for (int d = 0; d < DF; ++d) { ex[d] = __expf(slog[r * DF + d] - m); sum += ex[d]; }
            const float inv = __builtin_amdgcn_rcpf(sum);
            float* op = out + ((size_t)(row0 + r) * TT + t) * DF;
#pragma unroll
            for (int d = 0; d < DF; ++d) op[d] = ex[d] * inv;
        }
        // no barrier needed here: next conflicting write (slog in E) is fenced by 4 barriers
    }
}

extern "C" void kernel_launch(void* const* d_in, const int* in_sizes, int n_in,
                              void* d_out, int out_size, void* d_ws, size_t ws_size,
                              hipStream_t stream) {
    const float* input  = (const float*)d_in[0];
    const float* task   = (const float*)d_in[1];
    const float* W_in   = (const float*)d_in[2];
    const float* b_in   = (const float*)d_in[3];
    const float* W_task = (const float*)d_in[4];
    const float* b_task = (const float*)d_in[5];
    const float* W_x    = (const float*)d_in[6];
    const float* W_h    = (const float*)d_in[7];
    const float* b_lstm = (const float*)d_in[8];
    const float* W_out  = (const float*)d_in[9];
    const float* b_out  = (const float*)d_in[10];
    float* outp = (float*)d_out;

    lstm_fused_kernel<<<dim3(B_ / 16), dim3(NTHREADS), 0, stream>>>(
        input, task, W_in, b_in, W_task, b_task, W_x, W_h, b_lstm, W_out, b_out, outp);
}
